// Head_49443663511607
// MI455X (gfx1250) — matrix operand order repaired and checked by reading, not compiled, not run on hardware
//
#include <hip/hip_runtime.h>
#include <hip/hip_bf16.h>

typedef __bf16 bf16;
typedef __attribute__((ext_vector_type(8)))  __bf16 v8bf;
typedef __attribute__((ext_vector_type(16))) __bf16 v16bf;
typedef __attribute__((ext_vector_type(8)))  float  v8f;

constexpr int kT  = 256;   // sequence length
constexpr int kC  = 256;   // embedding dim
constexpr int kHS = 32;    // head size

__device__ __forceinline__ v8f wmma_bf16(v16bf a, v16bf b, v8f c) {
  // (neg_a, A, neg_b, B, c_mod, C, reuse_a, reuse_b)
  return __builtin_amdgcn_wmma_f32_16x16x32_bf16(false, a, false, b, (short)0, c,
                                                 false, false);
}

// Load a 16-element bf16 fragment from two contiguous 16-byte groups.
__device__ __forceinline__ v16bf load2x8(const bf16* p0, const bf16* p1) {
  const v8bf lo = *(const v8bf*)p0;
  const v8bf hi = *(const v8bf*)p1;
  return __builtin_shufflevector(lo, hi, 0, 1, 2, 3, 4, 5, 6, 7,
                                 8, 9, 10, 11, 12, 13, 14, 15);
}

// Reductions across each 16-lane half of the wave32 (rows of a WMMA D tile
// live entirely within one 16-lane half).
__device__ __forceinline__ float hmax16(float v) {
#pragma unroll
  for (int off = 1; off < 16; off <<= 1) v = fmaxf(v, __shfl_xor(v, off, 16));
  return v;
}
__device__ __forceinline__ float hsum16(float v) {
#pragma unroll
  for (int off = 1; off < 16; off <<= 1) v += __shfl_xor(v, off, 16);
  return v;
}

__launch_bounds__(128)
__global__ void attn_head_kernel(const float* __restrict__ x,
                                 const float* __restrict__ Wq,
                                 const float* __restrict__ Wk,
                                 const float* __restrict__ Wv,
                                 float* __restrict__ out) {
  __shared__ __attribute__((aligned(16))) bf16 qs[kT * kHS];      // [t][h] 16 KB
  __shared__ __attribute__((aligned(16))) bf16 ksh[kT * kHS];     // [t][h] 16 KB
  __shared__ __attribute__((aligned(16))) bf16 vT[kHS * kT];      // [h][t] 16 KB
  __shared__ __attribute__((aligned(16))) bf16 wT[3][kHS * kC];   // W^T[h][c] 48 KB
  __shared__ __attribute__((aligned(16))) bf16 ps[4][16 * 32];    // 4 KB restage

  const int b    = blockIdx.x;
  const int tid  = threadIdx.x;
  const int wave = tid >> 5;
  const int lane = tid & 31;
  const int hh   = lane >> 4;   // which 16-lane half
  const int l16  = lane & 15;

  // Disjoint, causally load-balanced M-tile partition: every wave gets
  // exactly 18 key-tile pairs in phase 2 (sum of mt/2+1 == 18 for each wave).
  const int mts[4] = {wave, 7 - wave, 8 + wave, 15 - wave};

  const float* xb = x + (size_t)b * kT * kC;

  // ---- Stage W^T (bf16) into LDS: coalesced float4 reads, transposed writes.
  {
    const float* Ws[3] = {Wq, Wk, Wv};
#pragma unroll
    for (int w = 0; w < 3; ++w) {
      const float* W = Ws[w];
      for (int i = tid; i < (kC * kHS) / 4; i += 128) {
        const float4 f = ((const float4*)W)[i];
        const int k = (i * 4) / kHS;   // row in W (= c)
        const int n = (i * 4) % kHS;   // col in W (= h)
        wT[w][(n + 0) * kC + k] = (bf16)f.x;
        wT[w][(n + 1) * kC + k] = (bf16)f.y;
        wT[w][(n + 2) * kC + k] = (bf16)f.z;
        wT[w][(n + 3) * kC + k] = (bf16)f.w;
      }
    }
  }
  __syncthreads();

  // ================= Phase 1: q/k/v = x @ W (bf16 WMMA, fp32 accum) ========
  for (int mi = 0; mi < 4; ++mi) {
    const int mt  = mts[mi];                // 16-row tile index
    const int row = mt * 16 + l16;          // this lane's A-matrix row
    v8f aq0 = {}, aq1 = {}, ak0 = {}, ak1 = {}, av0 = {}, av1 = {};
    for (int kk = 0; kk < 8; ++kk) {        // K loop over C in steps of 32
      const int K0 = kk * 32;
      // A fragment: x[row, K0..K0+31] in 16-bit A layout (fp32 -> bf16)
      v16bf afrag;
      const float* g0 = xb + row * kC + K0 + 16 * hh;  // K {0..7}/{16..23}
      const float* g1 = g0 + 8;                         // K {8..15}/{24..31}
#pragma unroll
      for (int e = 0; e < 8; ++e) {
        afrag[e]     = (bf16)g0[e];
        afrag[8 + e] = (bf16)g1[e];
      }
      // B fragments from W^T in LDS: 16 consecutive bf16 per lane.
      const int kb = K0 + 16 * hh;
      const bf16* c0q = &wT[0][l16 * kC + kb];
      const bf16* c1q = &wT[0][(16 + l16) * kC + kb];
      const bf16* c0k = &wT[1][l16 * kC + kb];
      const bf16* c1k = &wT[1][(16 + l16) * kC + kb];
      const bf16* c0v = &wT[2][l16 * kC + kb];
      const bf16* c1v = &wT[2][(16 + l16) * kC + kb];
      const v16bf bq0 = load2x8(c0q, c0q + 8);
      const v16bf bq1 = load2x8(c1q, c1q + 8);
      const v16bf bk0 = load2x8(c0k, c0k + 8);
      const v16bf bk1 = load2x8(c1k, c1k + 8);
      const v16bf bv0 = load2x8(c0v, c0v + 8);
      const v16bf bv1 = load2x8(c1v, c1v + 8);
      aq0 = wmma_bf16(afrag, bq0, aq0);
      aq1 = wmma_bf16(afrag, bq1, aq1);
      ak0 = wmma_bf16(afrag, bk0, ak0);
      ak1 = wmma_bf16(afrag, bk1, ak1);
      av0 = wmma_bf16(afrag, bv0, av0);
      av1 = wmma_bf16(afrag, bv1, av1);
    }
    // D layout: VGPR r holds row r (lanes 0-15) / row r+8 (lanes 16-31), col=l16
#pragma unroll
    for (int r = 0; r < 8; ++r) {
      const int t = mt * 16 + r + 8 * hh;
      qs [t * kHS + l16]       = (bf16)aq0[r];
      qs [t * kHS + 16 + l16]  = (bf16)aq1[r];
      ksh[t * kHS + l16]       = (bf16)ak0[r];
      ksh[t * kHS + 16 + l16]  = (bf16)ak1[r];
      vT [l16 * kT + t]        = (bf16)av0[r];   // transposed: [h][t]
      vT [(16 + l16) * kT + t] = (bf16)av1[r];
    }
  }
  __syncthreads();

  // ================= Phase 2: causal attention (online softmax) ============
  const float scale = 0.17677669529663687f;  // 1/sqrt(32)
  float* ob = out + (size_t)b * kT * kHS;

  for (int mi = 0; mi < 4; ++mi) {
    const int mt = mts[mi];
    // Q fragment in A layout: K dim = head dim (32)
    const bf16* q0 = qs + (mt * 16 + l16) * kHS + 16 * hh;
    const v16bf qa = load2x8(q0, q0 + 8);

    float mrow[8], lrow[8];
#pragma unroll
    for (int r = 0; r < 8; ++r) { mrow[r] = -3.0e38f; lrow[r] = 0.0f; }
    v8f o0 = {}, o1 = {};

    // Process key tiles in pairs so P.V runs with a full K=32 WMMA.
    // A pair's second tile past the diagonal is fully masked (P=0) -> no
    // wave-divergent branch around any WMMA (EXEC stays all-ones).
    const int npairs = mt / 2 + 1;
    for (int pt = 0; pt < npairs; ++pt) {
      const int nt0 = 2 * pt, nt1 = 2 * pt + 1;
      // K^T fragments in B layout: col n = key index, K dim = head dim
      const bf16* k0 = ksh + (nt0 * 16 + l16) * kHS + 16 * hh;
      const bf16* k1 = ksh + (nt1 * 16 + l16) * kHS + 16 * hh;
      const v16bf kb0 = load2x8(k0, k0 + 8);
      const v16bf kb1 = load2x8(k1, k1 + 8);
      v8f s0 = {}, s1 = {};
      s0 = wmma_bf16(qa, kb0, s0);
      s1 = wmma_bf16(qa, kb1, s1);

      const int gs0 = nt0 * 16 + l16, gs1 = nt1 * 16 + l16;
#pragma unroll
      for (int r = 0; r < 8; ++r) {
        const int gm = mt * 16 + r + 8 * hh;
        float a0 = s0[r] * scale; if (gs0 > gm) a0 = -3.0e38f;
        float a1 = s1[r] * scale; if (gs1 > gm) a1 = -3.0e38f;
        const float mx   = hmax16(fmaxf(a0, a1));
        const float mnew = fmaxf(mrow[r], mx);      // finite after first pair
        const float p0    = __expf(a0 - mnew);      // masked -> exp(-inf)=0
        const float p1    = __expf(a1 - mnew);
        const float alpha = __expf(mrow[r] - mnew);
        mrow[r] = mnew;
        lrow[r] = lrow[r] * alpha + hsum16(p0 + p1);
        o0[r] *= alpha;
        o1[r] *= alpha;
        // restage P row-major (16x32) so we can reload it in A layout
        bf16* pr = &ps[wave][(r + 8 * hh) * 32];
        pr[l16]      = (bf16)p0;
        pr[16 + l16] = (bf16)p1;
      }
      // P fragment in A layout (same-wave LDS RAW: ordered via DScnt)
      const bf16* pp = &ps[wave][l16 * 32 + 16 * hh];
      const v16bf pa = load2x8(pp, pp + 8);

      // V fragments in B layout from V^T: 16 consecutive sequence positions.
      const int srow = nt0 * 16 + 16 * hh;
      const bf16* v0 = &vT[l16 * kT + srow];
      const bf16* v1 = &vT[(16 + l16) * kT + srow];
      const v16bf vb0 = load2x8(v0, v0 + 8);
      const v16bf vb1 = load2x8(v1, v1 + 8);
      o0 = wmma_bf16(pa, vb0, o0);
      o1 = wmma_bf16(pa, vb1, o1);
    }
    // Normalize and write out (fp32)
#pragma unroll
    for (int r = 0; r < 8; ++r) {
      const float inv = 1.0f / lrow[r];
      const int t = mt * 16 + r + 8 * hh;
      ob[t * kHS + l16]      = o0[r] * inv;
      ob[t * kHS + 16 + l16] = o1[r] * inv;
    }
  }
}

extern "C" void kernel_launch(void* const* d_in, const int* in_sizes, int n_in,
                              void* d_out, int out_size, void* d_ws, size_t ws_size,
                              hipStream_t stream) {
  const float* x  = (const float*)d_in[0];
  const float* Wq = (const float*)d_in[1];
  const float* Wk = (const float*)d_in[2];
  const float* Wv = (const float*)d_in[3];
  float* out = (float*)d_out;
  const int B = in_sizes[0] / (kT * kC);   // 2048
  attn_head_kernel<<<dim3(B), dim3(128), 0, stream>>>(x, Wq, Wk, Wv, out);
}
